// DynamicReductionNetwork_8950711845773
// MI455X (gfx1250) — matrix operand order repaired and checked by reading, not compiled, not run on hardware
//
#include <hip/hip_runtime.h>
#include <math.h>

// ---------------- problem constants (from reference setup_inputs) ------------
#define Gn   32
#define Nn   1024
#define Dd   4
#define Hd   64
#define Kk   16
#define ADJW (Nn / 32)          // 32 u32 words per adjacency row
static const float FINF = 3.0e38f;

typedef _Float16 half_t;
typedef __attribute__((ext_vector_type(16))) _Float16 v16h;
typedef __attribute__((ext_vector_type(8)))  float    v8f;

// ---------------- WMMA helpers (CDNA5, wave32) -------------------------------
__device__ inline v8f wmma_f16(v16h a, v16h b, v8f c) {
  // D = A(16x32 f16) * B(32x16 f16) + C(16x16 f32)
  return __builtin_amdgcn_wmma_f32_16x16x32_f16(
      /*neg_a=*/false, a, /*neg_b=*/false, b,
      /*c_mod=*/(short)0, c, /*reuse_a=*/false, /*reuse_b=*/false);
}

// A-matrix 16x32 f16 fragment from row-major src (row stride ld halves).
// Layout: lanes 0-15 row M=lane, K=0..15; lanes 16-31 same
// rows, K=16..31; VGPR v holds 2 consecutive K values.
__device__ inline v16h load_a16(const half_t* base, int ld, int lane) {
  int row  = lane & 15;
  int koff = (lane >> 4) << 4;            // 0 or 16
  const half_t* p = base + row * ld + koff;
  v16h f;
#pragma unroll
  for (int v = 0; v < 8; ++v) {
    int k = 2 * v;  // 0,2,4,...,14
    f[2 * v]     = p[k];
    f[2 * v + 1] = p[k + 1];
  }
  return f;
}

// B-matrix 32x16 f16 fragment from row-major weights W[k][n] (row stride ld).
// lanes 0-15: col N=lane, K=0..15; lanes 16-31: col N=lane-16, K=16..31.
__device__ inline v16h load_b16(const half_t* base, int ld, int lane) {
  int n  = lane & 15;
  int kg = (lane >> 4) << 4;              // 0 or 16
  const half_t* p = base + kg * ld + n;
  v16h f;
#pragma unroll
  for (int v = 0; v < 8; ++v) {
    f[2 * v]     = p[(2 * v) * ld];
    f[2 * v + 1] = p[(2 * v + 1) * ld];
  }
  return f;
}

// B = X^T where X is row-major [n][ld]: B(k,n) = X[n*ld + k].
// Per lane this is 16 contiguous halves -> wide vector loads.
__device__ inline v16h load_bt16(const half_t* base, int ld, int lane) {
  int n  = lane & 15;
  int kg = (lane >> 4) << 4;
  const half_t* p = base + n * ld + kg;
  v16h f;
#pragma unroll
  for (int v = 0; v < 16; ++v) f[v] = p[v];
  return f;
}

__device__ inline float eluf(float x) { return x > 0.0f ? x : (expf(x) - 1.0f); }

// ---------------- 0) weight prep: f32 -> f16 (and W1 split/diff) ------------
__global__ void k_wprep(const float* __restrict__ W1,   // [128][128]
                        half_t* __restrict__ Wd,        // [64][128] top-bot
                        half_t* __restrict__ Wb,        // [64][128] bot
                        const float* __restrict__ W2,   // [128][64]
                        half_t* __restrict__ W2f) {     // [128][64]
  int t = blockIdx.x * blockDim.x + threadIdx.x;
  if (t < 64 * 128) {
    int k = t >> 7, o = t & 127;
    float top = W1[k * 128 + o], bot = W1[(64 + k) * 128 + o];
    Wd[t] = (half_t)(top - bot);
    Wb[t] = (half_t)bot;
  }
  int u = t - 64 * 128;
  if (u >= 0 && u < 128 * 64) W2f[u] = (half_t)W2[u];
}

// ---------------- 1) inputnet: h = ELU((datanorm*x) @ W_in + b_in) ----------
__global__ void k_input(const float* __restrict__ x, const float* __restrict__ dn,
                        const float* __restrict__ Win, const float* __restrict__ bin,
                        float* __restrict__ h, int* __restrict__ valid) {
  int gi = blockIdx.x * blockDim.x + threadIdx.x;
  if (gi >= Gn * Nn) return;
  const float* xp = x + (size_t)gi * Dd;
  float xi[Dd];
#pragma unroll
  for (int d = 0; d < Dd; ++d) xi[d] = xp[d] * dn[d];
  float* out = h + (size_t)gi * Hd;
  for (int c = 0; c < Hd; ++c) {
    float s = bin[c];
#pragma unroll
    for (int d = 0; d < Dd; ++d) s += xi[d] * Win[d * Hd + c];
    out[c] = eluf(s);
  }
  valid[gi] = 1;
}

// ---------------- 2) per-layer prep: f16 copy + ||h||^2 (from f16 values) ---
__global__ void k_prep(const float* __restrict__ h, half_t* __restrict__ hf,
                       float* __restrict__ sq) {
  int gi = blockIdx.x * blockDim.x + threadIdx.x;
  if (gi >= Gn * Nn) return;
  const float* p = h + (size_t)gi * Hd;
  half_t* q = hf + (size_t)gi * Hd;
  float s = 0.0f;
  for (int c = 0; c < Hd; ++c) {
    half_t hv = (half_t)p[c];
    q[c] = hv;
    float fv = (float)hv;
    s += fv * fv;
  }
  sq[gi] = s;
}

// ---------------- 3) kNN (WMMA dot tiles) + top-16 -> adjacency bitmask -----
// One wave per 16-row tile. d2(i,j) = sq_i + sq_j - 2*dot(h_i,h_j); invalid
// cols masked to +inf (reference masks cols only); invalid rows skipped (their
// edges die under emask anyway). Symmetrize bits => to_undirected+coalesce.
__global__ __launch_bounds__(32) void k_dist_topk(
    const half_t* __restrict__ hf, const float* __restrict__ sq,
    const int* __restrict__ valid, unsigned* __restrict__ adj) {
  __shared__ float drow[16 * Nn];   // 64 KB of the 320 KB WGP LDS
  int g = blockIdx.x >> 6;          // 64 row-tiles per graph
  int t = blockIdx.x & 63;
  int r0 = t * 16;
  int lane = threadIdx.x;
  const half_t*  hg   = hf    + (size_t)g * Nn * Hd;
  const float*   sqg  = sq    + (size_t)g * Nn;
  const int*     vg   = valid + (size_t)g * Nn;
  unsigned*      adjg = adj   + (size_t)g * Nn * ADJW;

  v16h a0 = load_a16(hg + (size_t)r0 * Hd,      Hd, lane);  // K 0..31
  v16h a1 = load_a16(hg + (size_t)r0 * Hd + 32, Hd, lane);  // K 32..63
  int nidx = lane & 15, mb = (lane >> 4) * 8;

  for (int ct = 0; ct < 64; ++ct) {
    int c0 = ct * 16;
    v16h b0 = load_bt16(hg + (size_t)c0 * Hd,      Hd, lane);
    v16h b1 = load_bt16(hg + (size_t)c0 * Hd + 32, Hd, lane);
    v8f acc = {};
    acc = wmma_f16(a0, b0, acc);
    acc = wmma_f16(a1, b1, acc);
    int c = c0 + nidx;
    float sqc = sqg[c];
    bool  vc  = vg[c] != 0;
#pragma unroll
    for (int r = 0; r < 8; ++r) {
      int m = mb + r;
      float d2 = sqg[r0 + m] + sqc - 2.0f * acc[r];
      drow[m * Nn + c] = vc ? d2 : FINF;
    }
  }
  __syncthreads();

  // top-K per row; ties -> lowest index (matches lax.top_k stability)
  for (int m = 0; m < 16; ++m) {
    int i = r0 + m;
    if (!vg[i]) continue;                 // uniform across wave
    float* row = drow + m * Nn;
    for (int it = 0; it < Kk; ++it) {
      float bv = FINF; int bj = Nn;
      for (int jj = lane; jj < Nn; jj += 32) {
        float v = row[jj];
        if (v < bv || (v == bv && jj < bj)) { bv = v; bj = jj; }
      }
#pragma unroll
      for (int off = 16; off >= 1; off >>= 1) {
        float ov = __shfl_xor(bv, off, 32);
        int   oj = __shfl_xor(bj, off, 32);
        if (ov < bv || (ov == bv && oj < bj)) { bv = ov; bj = oj; }
      }
      if (bv < FINF) {
        if (lane == 0) {
          atomicOr(&adjg[(size_t)i * ADJW + (bj >> 5)], 1u << (bj & 31));
          atomicOr(&adjg[(size_t)bj * ADJW + (i >> 5)], 1u << (i & 31));
          row[bj] = FINF;                 // remove from further rounds
        }
      }
      __syncthreads();
    }
  }
}

// ---------------- 4) node-level first-layer linears (WMMA GEMM) -------------
// A1 = h @ (W1_top - W1_bot) + b1 ; B1 = h @ W1_bot
__global__ __launch_bounds__(32) void k_nodelin(
    const half_t* __restrict__ hf, const half_t* __restrict__ Wd,
    const half_t* __restrict__ Wb, const float* __restrict__ b1,
    float* __restrict__ A1, float* __restrict__ B1) {
  __shared__ half_t arow[16 * Hd];
  int r0 = blockIdx.x * 16;          // over G*N rows
  int lane = threadIdx.x;
  for (int idx = lane; idx < 16 * Hd; idx += 32)
    arow[idx] = hf[(size_t)(r0 + (idx >> 6)) * Hd + (idx & 63)];
  __syncthreads();
  v16h a0 = load_a16(arow,      Hd, lane);
  v16h a1 = load_a16(arow + 32, Hd, lane);
  int nidx = lane & 15, mb = (lane >> 4) * 8;
  for (int w = 0; w < 2; ++w) {
    const half_t* W = w ? Wb : Wd;
    float* OUT = w ? B1 : A1;
#pragma unroll 1
    for (int nt = 0; nt < 8; ++nt) {
      v16h bf0 = load_b16(W + nt * 16,            128, lane);  // K 0..31
      v16h bf1 = load_b16(W + 32 * 128 + nt * 16, 128, lane);  // K 32..63
      v8f acc = {};
      acc = wmma_f16(a0, bf0, acc);
      acc = wmma_f16(a1, bf1, acc);
      int col = nt * 16 + nidx;
      float bias = w ? 0.0f : b1[col];
#pragma unroll
      for (int r = 0; r < 8; ++r)
        OUT[(size_t)(r0 + mb + r) * 128 + col] = acc[r] + bias;
    }
  }
}

// ---------------- 5) EdgeConv second layer + add-aggregation ----------------
// Per node i: gather neighbor set from bitmask, chunks of 16 edges:
//   A = f16(ELU(a_i + b_j))  [16,128];  M = ELU(A @ W2 + b2)  [16,64]
//   xn_i = sum over valid edge rows of M.
__global__ __launch_bounds__(32) void k_edgeconv(
    const float* __restrict__ A1, const float* __restrict__ B1,
    const half_t* __restrict__ W2f, const float* __restrict__ b2,
    const unsigned* __restrict__ adj, const int* __restrict__ valid,
    float* __restrict__ xn) {
  __shared__ half_t w2l[128 * 64];        // 16 KB, hot weight tile
  __shared__ half_t atile[16 * 128];      // 4 KB
  __shared__ unsigned short nlist[Nn];
  __shared__ float xacc[64];
  __shared__ int scnt;
  int g = blockIdx.x >> 10;
  int i = blockIdx.x & 1023;
  int lane = threadIdx.x;
  float* xni = xn + ((size_t)g * Nn + i) * Hd;
  if (!valid[g * Nn + i]) {               // uniform branch
    for (int c = lane; c < Hd; c += 32) xni[c] = 0.0f;
    return;
  }
  for (int idx = lane; idx < 128 * 64; idx += 32) w2l[idx] = W2f[idx];
  for (int c = lane; c < 64; c += 32) xacc[c] = 0.0f;
  if (lane == 0) {
    int cnt = 0;
    const unsigned* row = adj + ((size_t)g * Nn + i) * ADJW;
    for (int w = 0; w < ADJW; ++w) {
      unsigned bits = row[w];
      while (bits) {
        int b = __ffs(bits) - 1;
        bits &= bits - 1;
        nlist[cnt++] = (unsigned short)(w * 32 + b);
      }
    }
    scnt = cnt;
  }
  __syncthreads();
  int cnt = scnt;
  const float* a1i = A1 + ((size_t)g * Nn + i) * 128;
  int nch = (cnt + 15) >> 4;
  int nidx = lane & 15, mb = (lane >> 4) * 8;
  for (int ch = 0; ch < nch; ++ch) {
    // prefetch next chunk's B1 rows (global_prefetch_b8)
    int ne = ch * 16 + 16 + lane;
    if (ne < cnt) __builtin_prefetch(B1 + ((size_t)g * Nn + nlist[ne]) * 128, 0, 3);
    for (int idx = lane; idx < 16 * 128; idx += 32) {
      int e = idx >> 7, k = idx & 127;
      int ei = ch * 16 + e;
      float v = 0.0f;
      if (ei < cnt) {
        int j = nlist[ei];
        float p = a1i[k] + B1[((size_t)g * Nn + j) * 128 + k];
        v = eluf(p);
      }
      atile[idx] = (half_t)v;
    }
    __syncthreads();
#pragma unroll 1
    for (int nt = 0; nt < 4; ++nt) {
      v8f acc = {};
#pragma unroll
      for (int kc = 0; kc < 4; ++kc) {
        v16h af = load_a16(atile + kc * 32, 128, lane);
        v16h bf = load_b16(w2l + kc * 32 * 64 + nt * 16, 64, lane);
        acc = wmma_f16(af, bf, acc);
      }
      int col = nt * 16 + nidx;
      float bias = b2[col];
      float s = 0.0f;
#pragma unroll
      for (int r = 0; r < 8; ++r) {
        int e = ch * 16 + mb + r;
        float v = eluf(acc[r] + bias);
        if (e < cnt) s += v;              // mask padded edges (msg = 0)
      }
      s += __shfl_xor(s, 16, 32);         // combine the two half-lane groups
      if (lane < 16) xacc[col] += s;
    }
    __syncthreads();
  }
  for (int c = lane; c < Hd; c += 32) xni[c] = xacc[c];
}

// ---------------- 6) degree -> inv (normalized cut weights) -----------------
__global__ void k_deg(const unsigned* __restrict__ adj, float* __restrict__ inv) {
  int gi = blockIdx.x * blockDim.x + threadIdx.x;
  if (gi >= Gn * Nn) return;
  const unsigned* row = adj + (size_t)gi * ADJW;
  int d = 0;
  for (int w = 0; w < ADJW; ++w) d += __popc(row[w]);
  inv[gi] = (d > 0) ? 1.0f / (float)d : 0.0f;
}

// ---------------- 7) graclus: per-node best (heaviest) neighbor -------------
// bcol = segment_max(cols where w==bw): ties -> LARGEST col index.
__global__ __launch_bounds__(32) void k_graclus(
    const float* __restrict__ xn, const unsigned* __restrict__ adj,
    const float* __restrict__ inv, const int* __restrict__ valid,
    int* __restrict__ bcol) {
  __shared__ float xi[Hd];
  int g = blockIdx.x >> 10;
  int i = blockIdx.x & 1023;
  int lane = threadIdx.x;
  int gi = g * Nn + i;
  if (!valid[gi]) {
    if (lane == 0) bcol[gi] = -1;
    return;
  }
  const float* xrow = xn + (size_t)gi * Hd;
  for (int c = lane; c < Hd; c += 32) xi[c] = xrow[c];
  __syncthreads();
  float invi = inv[gi];
  unsigned bits = adj[(size_t)gi * ADJW + lane];  // lane <-> word (32 words)
  float bw = -1e30f; int bj = -1;
  while (bits) {
    int b = __ffs(bits) - 1;
    bits &= bits - 1;
    int j = lane * 32 + b;
    if (j == i) continue;                 // mmask excludes self loops
    const float* xj = xn + ((size_t)g * Nn + j) * Hd;
    float s = 0.0f;
    for (int c = 0; c < Hd; ++c) { float d = xi[c] - xj[c]; s += d * d; }
    float w = sqrtf(s + 1e-12f) * (invi + inv[g * Nn + j]);
    if (w > bw || (w == bw && j > bj)) { bw = w; bj = j; }
  }
#pragma unroll
  for (int off = 16; off >= 1; off >>= 1) {
    float ow = __shfl_xor(bw, off, 32);
    int   oj = __shfl_xor(bj, off, 32);
    if (ow > bw || (ow == bw && oj > bj)) { bw = ow; bj = oj; }
  }
  if (lane == 0) bcol[gi] = bj;
}

// ---------------- 8) mutual matching + max_pool -----------------------------
__global__ void k_pool(const float* __restrict__ xn, const int* __restrict__ bcol,
                       const int* __restrict__ valid, float* __restrict__ hnew,
                       int* __restrict__ vnew) {
  int gi = blockIdx.x * blockDim.x + threadIdx.x;
  if (gi >= Gn * Nn) return;
  int g = gi >> 10, i = gi & 1023;
  int b = bcol[gi];
  int v = valid[gi];
  float* out = hnew + (size_t)gi * Hd;
  if (!v) {
    vnew[gi] = 0;
    for (int c = 0; c < Hd; ++c) out[c] = 0.0f;
    return;
  }
  bool mut = (b >= 0) && (bcol[g * Nn + b] == i);
  const float* xi = xn + (size_t)gi * Hd;
  if (mut && b < i) {                     // absorbed into representative b
    vnew[gi] = 0;
    for (int c = 0; c < Hd; ++c) out[c] = 0.0f;
    return;
  }
  vnew[gi] = 1;
  if (mut) {
    const float* xj = xn + ((size_t)(g * Nn + b)) * Hd;
    for (int c = 0; c < Hd; ++c) out[c] = fmaxf(xi[c], xj[c]);
  } else {
    for (int c = 0; c < Hd; ++c) out[c] = xi[c];
  }
}

// ---------------- 9) global max pool + output MLP ---------------------------
__global__ __launch_bounds__(64) void k_out(
    const float* __restrict__ h, const int* __restrict__ valid,
    const float* __restrict__ Wo1, const float* __restrict__ bo1,
    const float* __restrict__ Wo2, const float* __restrict__ bo2,
    const float* __restrict__ Wo3, const float* __restrict__ bo3,
    float* __restrict__ out) {
  __shared__ float gfeat[64], t1[64], t2[64], red[64];
  int g = blockIdx.x;
  int c = threadIdx.x;
  float m = -1e9f;                        // NEG sentinel, matches reference
  for (int i = 0; i < Nn; ++i)
    if (valid[g * Nn + i]) m = fmaxf(m, h[((size_t)g * Nn + i) * Hd + c]);
  gfeat[c] = m;
  __syncthreads();
  float s = bo1[c];
  for (int k = 0; k < 64; ++k) s += gfeat[k] * Wo1[k * 64 + c];
  t1[c] = eluf(s);
  __syncthreads();
  s = bo2[c];
  for (int k = 0; k < 64; ++k) s += t1[k] * Wo2[k * 64 + c];
  t2[c] = eluf(s);
  __syncthreads();
  red[c] = t2[c] * Wo3[c];
  __syncthreads();
  for (int off = 32; off >= 1; off >>= 1) {
    if (c < off) red[c] += red[c + off];
    __syncthreads();
  }
  if (c == 0) out[g] = red[0] + bo3[0];
}

// ---------------- host: carve workspace, run pipeline -----------------------
extern "C" void kernel_launch(void* const* d_in, const int* in_sizes, int n_in,
                              void* d_out, int out_size, void* d_ws, size_t ws_size,
                              hipStream_t stream) {
  (void)in_sizes; (void)n_in; (void)out_size; (void)ws_size;
  const float* x   = (const float*)d_in[0];
  const float* dn  = (const float*)d_in[1];
  const float* Win = (const float*)d_in[2];
  const float* bin = (const float*)d_in[3];
  const float* W1a = (const float*)d_in[4];
  const float* b1a = (const float*)d_in[5];
  const float* W2a = (const float*)d_in[6];
  const float* b2a = (const float*)d_in[7];
  const float* W1b = (const float*)d_in[8];
  const float* b1b = (const float*)d_in[9];
  const float* W2b = (const float*)d_in[10];
  const float* b2b = (const float*)d_in[11];
  const float* Wo1 = (const float*)d_in[12];
  const float* bo1 = (const float*)d_in[13];
  const float* Wo2 = (const float*)d_in[14];
  const float* bo2 = (const float*)d_in[15];
  const float* Wo3 = (const float*)d_in[16];
  const float* bo3 = (const float*)d_in[17];
  float* out = (float*)d_out;

  const size_t GN = (size_t)Gn * Nn;
  size_t off = 0;
  auto carve = [&](size_t bytes) {
    void* p = (char*)d_ws + off;
    off = (off + bytes + 255) & ~(size_t)255;
    return p;
  };
  float*    hA   = (float*)carve(GN * Hd * 4);
  float*    hB   = (float*)carve(GN * Hd * 4);
  half_t*   hf   = (half_t*)carve(GN * Hd * 2);
  float*    sq   = (float*)carve(GN * 4);
  int*      vA   = (int*)carve(GN * 4);
  int*      vB   = (int*)carve(GN * 4);
  unsigned* adj  = (unsigned*)carve(GN * ADJW * 4);
  float*    A1   = (float*)carve(GN * 128 * 4);
  float*    B1   = (float*)carve(GN * 128 * 4);
  float*    xnb  = (float*)carve(GN * Hd * 4);
  int*      bcol = (int*)carve(GN * 4);
  float*    inv  = (float*)carve(GN * 4);
  half_t*   WdA  = (half_t*)carve(64 * 128 * 2);
  half_t*   WbA  = (half_t*)carve(64 * 128 * 2);
  half_t*   W2fA = (half_t*)carve(128 * 64 * 2);
  half_t*   WdB  = (half_t*)carve(64 * 128 * 2);
  half_t*   WbB  = (half_t*)carve(64 * 128 * 2);
  half_t*   W2fB = (half_t*)carve(128 * 64 * 2);

  const int WT = 64 * 128 + 128 * 64;
  k_wprep<<<(WT + 255) / 256, 256, 0, stream>>>(W1a, WdA, WbA, W2a, W2fA);
  k_wprep<<<(WT + 255) / 256, 256, 0, stream>>>(W1b, WdB, WbB, W2b, W2fB);
  k_input<<<(int)((GN + 255) / 256), 256, 0, stream>>>(x, dn, Win, bin, hA, vA);

  // ---- layer A: hA/vA -> hB/vB
  k_prep<<<(int)((GN + 255) / 256), 256, 0, stream>>>(hA, hf, sq);
  (void)hipMemsetAsync(adj, 0, GN * ADJW * 4, stream);
  k_dist_topk<<<Gn * 64, 32, 0, stream>>>(hf, sq, vA, adj);
  k_nodelin<<<(int)(GN / 16), 32, 0, stream>>>(hf, WdA, WbA, b1a, A1, B1);
  k_edgeconv<<<(int)GN, 32, 0, stream>>>(A1, B1, W2fA, b2a, adj, vA, xnb);
  k_deg<<<(int)((GN + 255) / 256), 256, 0, stream>>>(adj, inv);
  k_graclus<<<(int)GN, 32, 0, stream>>>(xnb, adj, inv, vA, bcol);
  k_pool<<<(int)((GN + 255) / 256), 256, 0, stream>>>(xnb, bcol, vA, hB, vB);

  // ---- layer B: hB/vB -> hA/vA
  k_prep<<<(int)((GN + 255) / 256), 256, 0, stream>>>(hB, hf, sq);
  (void)hipMemsetAsync(adj, 0, GN * ADJW * 4, stream);
  k_dist_topk<<<Gn * 64, 32, 0, stream>>>(hf, sq, vB, adj);
  k_nodelin<<<(int)(GN / 16), 32, 0, stream>>>(hf, WdB, WbB, b1b, A1, B1);
  k_edgeconv<<<(int)GN, 32, 0, stream>>>(A1, B1, W2fB, b2b, adj, vB, xnb);
  k_deg<<<(int)((GN + 255) / 256), 256, 0, stream>>>(adj, inv);
  k_graclus<<<(int)GN, 32, 0, stream>>>(xnb, adj, inv, vB, bcol);
  k_pool<<<(int)((GN + 255) / 256), 256, 0, stream>>>(xnb, bcol, vB, hA, vA);

  // ---- global pool + MLP head
  k_out<<<Gn, 64, 0, stream>>>(hA, vA, Wo1, bo1, Wo2, bo2, Wo3, bo3, out);
}
